// WirelessCompressor_20753281974551
// MI455X (gfx1250) — matrix operand order repaired and checked by reading, not compiled, not run on hardware
//
#include <hip/hip_runtime.h>

typedef __attribute__((ext_vector_type(16))) __bf16 v16bf;
typedef __attribute__((ext_vector_type(8)))  __bf16 v8bf;
typedef __attribute__((ext_vector_type(8)))  float  v8f;

#define L_DIM      64
#define NCODE      1024
#define URA_DIM    256
#define ROW_STRIDE 272                      /* bytes: 64 hi bf16 + 64 lo bf16 + 16 pad */
#define Q2_OFF     (NCODE * ROW_STRIDE)     /* 278528 */
#define SMEM_BYTES (Q2_OFF + NCODE * 4)     /* 282624 B < 320 KB WGP LDS */

union V16U { v16bf v; v8bf h[2]; __bf16 e[16]; };

__global__ __launch_bounds__(256)
void wc_argmin_gather_kernel(const float* __restrict__ X,
                             const float* __restrict__ Q,
                             const float* __restrict__ C,
                             float* __restrict__ out,
                             int Nrows)
{
    extern __shared__ char smem[];
    const int tid  = threadIdx.x;
    const int lane = tid & 31;
    const int wave = tid >> 5;
    const int m    = lane & 15;   // row-in-tile (A) / col-in-tile (B)
    const int g    = lane >> 4;   // lane half

    // ---- Stage codebook into LDS as bf16 hi/lo + q2 (whole WG, uniform trip count) ----
    for (int c = tid; c < NCODE; c += 256) {
        const float4* qr4 = (const float4*)(Q + (size_t)c * L_DIM);
        __bf16* hi = (__bf16*)(smem + c * ROW_STRIDE);
        __bf16* lo = hi + 64;
        float s = 0.0f;
        #pragma unroll
        for (int k4 = 0; k4 < L_DIM / 4; ++k4) {
            float4 v4 = qr4[k4];
            float vv[4] = {v4.x, v4.y, v4.z, v4.w};
            #pragma unroll
            for (int j = 0; j < 4; ++j) {
                float v = vv[j];
                s = fmaf(v, v, s);
                __bf16 h = (__bf16)v;
                hi[4 * k4 + j] = h;
                lo[4 * k4 + j] = (__bf16)(v - (float)h);
            }
        }
        ((float*)(smem + Q2_OFF))[c] = s;
    }
    __syncthreads();

    // ---- Load this wave's 16x64 A tile as bf16 hi/lo (two K=32 chunks) ----
    // A layout (16-bit 16x32): lane(m,g) elems 0..7 = K[8g..8g+8), elems 8..15 = K[16+8g..16+8g+8)
    const int rowBase = blockIdx.x * 128 + wave * 16;
    const float* xr = X + (size_t)(rowBase + m) * L_DIM;
    V16U Ah[2], Al[2];
    #pragma unroll
    for (int c2 = 0; c2 < 2; ++c2) {
        #pragma unroll
        for (int h2 = 0; h2 < 2; ++h2) {
            int k0 = 32 * c2 + 8 * g + 16 * h2;
            #pragma unroll
            for (int j = 0; j < 8; ++j) {
                float v = xr[k0 + j];
                __bf16 h = (__bf16)v;
                Ah[c2].e[8 * h2 + j] = h;
                Al[c2].e[8 * h2 + j] = (__bf16)(v - (float)h);
            }
        }
    }

    // ---- Sweep 64 column tiles, 2 tiles per iteration (independent acc chains) ----
    float best[8];
    int   besti[8];
    #pragma unroll
    for (int j = 0; j < 8; ++j) { best[j] = 3.4e38f; besti[j] = 0; }

    const float* q2s = (const float*)(smem + Q2_OFF);

    for (int t = 0; t < NCODE / 16; t += 2) {
        v8f   acc[2];
        float q2v[2];
        int   col[2];
        V16U  Bh[2][2], Bl[2][2];

        // Load both tiles' B operands first so their WMMA chains fully overlap.
        #pragma unroll
        for (int u = 0; u < 2; ++u) {
            col[u] = (t + u) * 16 + m;
            const char* rowp = smem + col[u] * ROW_STRIDE;
            #pragma unroll
            for (int c2 = 0; c2 < 2; ++c2) {
                int koff = (32 * c2 + 8 * g) * 2;
                Bh[u][c2].h[0] = *(const v8bf*)(rowp + koff);
                Bh[u][c2].h[1] = *(const v8bf*)(rowp + koff + 32);
                Bl[u][c2].h[0] = *(const v8bf*)(rowp + 128 + koff);
                Bl[u][c2].h[1] = *(const v8bf*)(rowp + 128 + koff + 32);
            }
            q2v[u] = q2s[col[u]];
        }

        #pragma unroll
        for (int u = 0; u < 2; ++u) {
            v8f a = {};
            #pragma unroll
            for (int c2 = 0; c2 < 2; ++c2) {
                a = __builtin_amdgcn_wmma_f32_16x16x32_bf16(false, Ah[c2].v, false, Bh[u][c2].v,
                                                            (short)0, a, false, false);
                a = __builtin_amdgcn_wmma_f32_16x16x32_bf16(false, Ah[c2].v, false, Bl[u][c2].v,
                                                            (short)0, a, false, false);
                a = __builtin_amdgcn_wmma_f32_16x16x32_bf16(false, Al[c2].v, false, Bh[u][c2].v,
                                                            (short)0, a, false, false);
            }
            acc[u] = a;
        }

        #pragma unroll
        for (int u = 0; u < 2; ++u) {
            #pragma unroll
            for (int j = 0; j < 8; ++j) {
                float s = fmaf(-2.0f, acc[u][j], q2v[u]);   // x2[row] constant -> dropped
                if (s < best[j]) { best[j] = s; besti[j] = col[u]; }
            }
        }
    }

    // ---- Cross-lane argmin over the 16 columns sharing each row (within lane halves) ----
    #pragma unroll
    for (int mask = 1; mask <= 8; mask <<= 1) {
        #pragma unroll
        for (int j = 0; j < 8; ++j) {
            float ov = __shfl_xor(best[j],  mask, 32);
            int   oi = __shfl_xor(besti[j], mask, 32);
            if (ov < best[j] || (ov == best[j] && oi < besti[j])) {
                best[j] = ov; besti[j] = oi;
            }
        }
    }
    // Now every lane in half g holds the argmin for rows rowBase + 8g + j.

    float* idxOut = out;
    float* qOut   = out + (size_t)Nrows;
    float* uOut   = qOut + (size_t)Nrows * L_DIM;

    if (m == 0) {   // lanes 0 and 16
        #pragma unroll
        for (int j = 0; j < 8; ++j)
            idxOut[rowBase + 8 * g + j] = (float)besti[j];
    }

    // ---- Wave-cooperative gather of Q[idx] (64 f32) and C_syn[idx] (256 f32) ----
    #pragma unroll
    for (int r = 0; r < 16; ++r) {
        const int src = (r & 8) ? 16 : 0;
        const int idx = __shfl(besti[r & 7], src, 32);
        const int row = rowBase + r;

        const float4* qsrc = (const float4*)(Q + (size_t)idx * L_DIM);
        float4*       qdst = (float4*)(qOut + (size_t)row * L_DIM);
        if (lane < 16) qdst[lane] = qsrc[lane];

        const float4* usrc = (const float4*)(C + (size_t)idx * URA_DIM);
        float4*       udst = (float4*)(uOut + (size_t)row * URA_DIM);
        udst[lane]      = usrc[lane];
        udst[lane + 32] = usrc[lane + 32];
    }
}

extern "C" void kernel_launch(void* const* d_in, const int* in_sizes, int n_in,
                              void* d_out, int out_size, void* d_ws, size_t ws_size,
                              hipStream_t stream) {
    (void)n_in; (void)out_size; (void)d_ws; (void)ws_size;
    const float* X = (const float*)d_in[0];   // splits_flat [N,64]
    const float* Q = (const float*)d_in[1];   // Q_codebook [1024,64]
    const float* C = (const float*)d_in[2];   // C_syn      [1024,256]
    float* out = (float*)d_out;

    const int Nrows = in_sizes[0] / L_DIM;    // 131072

    hipFuncSetAttribute((const void*)wc_argmin_gather_kernel,
                        hipFuncAttributeMaxDynamicSharedMemorySize, SMEM_BYTES);

    dim3 grid(Nrows / 128);
    dim3 block(256);
    wc_argmin_gather_kernel<<<grid, block, SMEM_BYTES, stream>>>(X, Q, C, out, Nrows);
}
